// XNorCosAttention_74758200754302
// MI455X (gfx1250) — compile-verified, matrix operand order checked by reading, not run on hardware
//
#include <hip/hip_runtime.h>
#include <hip/hip_bf16.h>

// ---------------------------------------------------------------------------
// XNorCosAttention (cosFormer linear attention) for MI455X / gfx1250, wave32.
// All GEMMs use v_wmma_f32_16x16x32_bf16 (bf16 in, fp32 accum).
// Big GEMMs: double-buffered LDS fed by GLOBAL_LOAD_ASYNC_TO_LDS_B128
// (ASYNCcnt) so global traffic overlaps WMMA on the current buffer.
// ---------------------------------------------------------------------------

typedef __bf16 bf16;
typedef __attribute__((ext_vector_type(4)))  bf16  v4bf;
typedef __attribute__((ext_vector_type(8)))  bf16  v8bf;
typedef __attribute__((ext_vector_type(16))) bf16  v16bf;
typedef __attribute__((ext_vector_type(8)))  float v8f;

// Problem constants (match reference)
constexpr int BB  = 8;
constexpr int LL  = 2048;
constexpr int EE  = 1024;
constexpr int HH  = 16;
constexpr int DD  = 64;          // head dim
constexpr int D2  = 128;         // 2*head dim (feature dim)
constexpr int TT  = BB * LL;     // 16384 tokens
constexpr int NBH = BB * HH;     // 128 (b,h) pairs
constexpr float EPSV = 1e-6f;
constexpr float PI_2 = 1.5707963267948966f;

// --- WMMA helpers -----------------------------------------------------------
// A 16x32 bf16 fragment (or B 32x16 pre-transposed): lane (r=l&15, h=l>>4)
// holds K = [8h..8h+7] and [16+8h..16+8h+7] of row r — two contiguous chunks.
__device__ __forceinline__ v16bf load_frag(const bf16* base) {
  v8bf lo = *(const v8bf*)(base);
  v8bf hi = *(const v8bf*)(base + 16);
  return __builtin_shufflevector(lo, hi, 0,1,2,3,4,5,6,7,8,9,10,11,12,13,14,15);
}
__device__ __forceinline__ v8f wmma_bf16(v16bf a, v16bf b, v8f c) {
  return __builtin_amdgcn_wmma_f32_16x16x32_bf16(false, a, false, b,
                                                 (short)0, c, false, false);
}

// --- CDNA5 async global->LDS copy (16B per lane, tracked by ASYNCcnt) -------
__device__ __forceinline__ void async_copy16(const bf16* g, bf16* l) {
  // Generic LDS address truncates to the 32-bit LDS byte offset.
  unsigned lds_off = (unsigned)(uintptr_t)l;
  asm volatile("global_load_async_to_lds_b128 %0, %1, off"
               :: "v"(lds_off), "v"(g) : "memory");
}
__device__ __forceinline__ void wait_async0() {
  asm volatile("s_wait_asynccnt 0x0" ::: "memory");
}

// --- K0a: transpose + convert weights: WT[n][k] = (bf16)W[k][n] -------------
__global__ __launch_bounds__(256) void transpose_w_kernel(
    const float* __restrict__ W, bf16* __restrict__ WT) {
  __shared__ float tile[32][33];
  const int x = threadIdx.x, y0 = threadIdx.y;         // 32 x 8
  const int bx = blockIdx.x * 32, by = blockIdx.y * 32;
#pragma unroll
  for (int i = 0; i < 4; ++i) {
    int y = y0 + i * 8;
    tile[y][x] = W[(long)(by + y) * EE + bx + x];
  }
  __syncthreads();
#pragma unroll
  for (int i = 0; i < 4; ++i) {
    int y = y0 + i * 8;
    WT[(long)(bx + y) * EE + by + x] = (bf16)tile[x][y];
  }
}

// --- K0b: fp32 -> bf16 bulk convert -----------------------------------------
__global__ __launch_bounds__(256) void convert_bf16_kernel(
    const float* __restrict__ x, bf16* __restrict__ y) {
  long i = ((long)blockIdx.x * blockDim.x + threadIdx.x) * 4;
  float4 f = *(const float4*)(x + i);
  v4bf o;
  o[0] = (bf16)f.x; o[1] = (bf16)f.y; o[2] = (bf16)f.z; o[3] = (bf16)f.w;
  *(v4bf*)(y + i) = o;
}

// --- Big NT GEMM: C(MxN) = A(MxK) * Bt(NxK)^T + bias ------------------------
// grid(N/128, M/128), 256 threads = 8 waves (2 Mwave x 4 Nwave),
// wave tile 64x32 = 4x2 WMMA tiles. Double-buffered 128x32 bf16 LDS tiles
// filled by async global->LDS b128 copies; stride 40 (20 dwords/row) gives
// conflict-free 16-row fragment reads across 64 banks.
template <bool OUT_BF16>
__global__ __launch_bounds__(256) void gemm_nt_kernel(
    const bf16* __restrict__ A, const bf16* __restrict__ Bt,
    const float* __restrict__ bias, void* __restrict__ Cout,
    const int K, const int N) {
  constexpr int BM = 128, BN = 128, BK = 32, LDSS = 40;
  __shared__ __align__(16) bf16 As[2][BM * LDSS];
  __shared__ __align__(16) bf16 Bs[2][BN * LDSS];

  const int t = threadIdx.x;
  const int wave = t >> 5, lane = t & 31;
  const int wm = wave >> 2, wn = wave & 3;
  const int r = lane & 15, h = lane >> 4;
  const int  blockN = blockIdx.x * BN;
  const long blockM = (long)blockIdx.y * BM;

  // Each thread owns two 16B chunks of each 128x32 tile (512 chunks total).
  const int row0 = t >> 2,          col0 = (t & 3) * 8;
  const int row1 = (t + 256) >> 2,  col1 = ((t + 256) & 3) * 8;
  const bf16* Arow0 = A + (blockM + row0) * K + col0;
  const bf16* Arow1 = A + (blockM + row1) * K + col1;
  const bf16* Brow0 = Bt + (long)(blockN + row0) * K + col0;
  const bf16* Brow1 = Bt + (long)(blockN + row1) * K + col1;

  auto issue = [&](int buf, int kk) {
    async_copy16(Arow0 + kk, &As[buf][row0 * LDSS + col0]);
    async_copy16(Arow1 + kk, &As[buf][row1 * LDSS + col1]);
    async_copy16(Brow0 + kk, &Bs[buf][row0 * LDSS + col0]);
    async_copy16(Brow1 + kk, &Bs[buf][row1 * LDSS + col1]);
  };

  issue(0, 0);
  wait_async0();
  __syncthreads();

  v8f acc[4][2] = {};
  const int nkb = K / BK;
  for (int kb = 0; kb < nkb; ++kb) {
    const int cur = kb & 1;
    if (kb + 1 < nkb) issue(cur ^ 1, (kb + 1) * BK);  // overlap with compute

    v16bf af[4], bfr[2];
#pragma unroll
    for (int i = 0; i < 4; ++i)
      af[i] = load_frag(&As[cur][(wm * 64 + i * 16 + r) * LDSS + 8 * h]);
#pragma unroll
    for (int j = 0; j < 2; ++j)
      bfr[j] = load_frag(&Bs[cur][(wn * 32 + j * 16 + r) * LDSS + 8 * h]);
#pragma unroll
    for (int i = 0; i < 4; ++i)
#pragma unroll
      for (int j = 0; j < 2; ++j)
        acc[i][j] = wmma_bf16(af[i], bfr[j], acc[i][j]);

    if (kb + 1 < nkb) wait_async0();   // next buffer resident before barrier
    __syncthreads();
  }

  // Epilogue. C layout: lane c=l&15 is column, vgpr v -> row (l>>4)*8 + v.
  const int cN = lane & 15, moff = (lane >> 4) * 8;
#pragma unroll
  for (int j = 0; j < 2; ++j) {
    const int n = blockN + wn * 32 + j * 16 + cN;
    const float bn = bias[n];
#pragma unroll
    for (int i = 0; i < 4; ++i)
#pragma unroll
      for (int v = 0; v < 8; ++v) {
        const long m = blockM + wm * 64 + i * 16 + moff + v;
        const float val = acc[i][j][v] + bn;
        if (OUT_BF16) ((bf16*)Cout)[m * N + n] = (bf16)val;
        else          ((float*)Cout)[m * N + n] = val;
      }
  }
}

// --- K2: per-head softmax + complement + cos/sin featurize ------------------
// One 64-thread block handles (b,h) x 64 consecutive l's. Emits:
//   QF/CQF  : (B,H,L,128) row-major (A operand of attn GEMM)
//   KFt/CKFt: (B,H,128,L) row-major (A operand of kv GEMM) via LDS transpose
//   Vt      : (B,H,64,L)  row-major (B operand of kv GEMM)
//   KSUM/CKSUM: (B,H,128) fp32 row sums (atomicAdd)
__global__ __launch_bounds__(64) void featurize_kernel(
    const bf16* __restrict__ Qbf, const bf16* __restrict__ Kbf,
    const bf16* __restrict__ Vbf,
    bf16* __restrict__ QF, bf16* __restrict__ CQF,
    bf16* __restrict__ KFt, bf16* __restrict__ CKFt, bf16* __restrict__ Vt,
    float* __restrict__ KSUM, float* __restrict__ CKSUM) {
  constexpr int LB = 64, SK = 130, SV = 66;       // padded LDS strides
  __shared__ __align__(16) bf16 kf_s[LB * SK];
  __shared__ __align__(16) bf16 ck_s[LB * SK];
  __shared__ __align__(16) bf16 v_s [LB * SV];

  const int tid = threadIdx.x;                    // 0..63
  const int lblock = blockIdx.x;                  // 0..L/64-1
  const int bh = blockIdx.y;                      // 0..127
  const int b = bh >> 4, hh = bh & 15;
  const int l = lblock * LB + tid;
  const long tok = (long)b * LL + l;

  const bf16* qrow = Qbf + tok * EE + hh * DD;
  const bf16* krow = Kbf + tok * EE + hh * DD;
  const bf16* vrow = Vbf + tok * EE + hh * DD;

  float q[DD], k[DD];
#pragma unroll
  for (int c8 = 0; c8 < 8; ++c8) {
    v8bf xq = *(const v8bf*)(qrow + c8 * 8);
    v8bf xk = *(const v8bf*)(krow + c8 * 8);
    v8bf xv = *(const v8bf*)(vrow + c8 * 8);
#pragma unroll
    for (int i = 0; i < 8; ++i) {
      q[c8 * 8 + i] = (float)xq[i];
      k[c8 * 8 + i] = (float)xk[i];
      v_s[tid * SV + c8 * 8 + i] = xv[i];
    }
  }

  // softmax over head dim (64), in fp32
  float mq = -1e30f, mk = -1e30f;
#pragma unroll
  for (int d = 0; d < DD; ++d) { mq = fmaxf(mq, q[d]); mk = fmaxf(mk, k[d]); }
  float sq = 0.f, sk = 0.f;
#pragma unroll
  for (int d = 0; d < DD; ++d) {
    q[d] = __expf(q[d] - mq); sq += q[d];
    k[d] = __expf(k[d] - mk); sk += k[d];
  }
  const float iq = 1.0f / sq, ik = 1.0f / sk;
#pragma unroll
  for (int d = 0; d < DD; ++d) { q[d] *= iq; k[d] *= ik; }

  const float ang = PI_2 * (float)(l + 1) / (float)LL;
  const float sv = __sinf(ang), cv = __cosf(ang);

  // QF / CQF rows (contiguous 128 bf16 each)
  const long qfb = ((long)bh * LL + l) * D2;
#pragma unroll
  for (int c8 = 0; c8 < 16; ++c8) {
    v8bf o1, o2;
#pragma unroll
    for (int i = 0; i < 8; ++i) {
      const int d = c8 * 8 + i;
      const float base = (d < DD) ? q[d] : q[d - DD];
      const float w = (d < DD) ? sv : cv;
      o1[i] = (bf16)(base * w);
      o2[i] = (bf16)((1.0f - base) * w);
    }
    *(v8bf*)(QF + qfb + c8 * 8) = o1;
    *(v8bf*)(CQF + qfb + c8 * 8) = o2;
  }

  // k features into LDS (l-major), then transposed coalesced store
#pragma unroll
  for (int d = 0; d < DD; ++d) {
    kf_s[tid * SK + d]      = (bf16)(k[d] * sv);
    kf_s[tid * SK + DD + d] = (bf16)(k[d] * cv);
    ck_s[tid * SK + d]      = (bf16)((1.0f - k[d]) * sv);
    ck_s[tid * SK + DD + d] = (bf16)((1.0f - k[d]) * cv);
  }
  __syncthreads();

#pragma unroll
  for (int rr = 0; rr < 2; ++rr) {
    const int dd = tid + rr * 64;
    float s1 = 0.f, s2 = 0.f;
    const long o1b = ((long)bh * D2 + dd) * LL + lblock * LB;
#pragma unroll
    for (int c8 = 0; c8 < 8; ++c8) {
      v8bf a, b2;
#pragma unroll
      for (int i = 0; i < 8; ++i) {
        const int ll = c8 * 8 + i;
        const bf16 xv = kf_s[ll * SK + dd]; a[i]  = xv; s1 += (float)xv;
        const bf16 yv = ck_s[ll * SK + dd]; b2[i] = yv; s2 += (float)yv;
      }
      *(v8bf*)(KFt + o1b + c8 * 8) = a;
      *(v8bf*)(CKFt + o1b + c8 * 8) = b2;
    }
    atomicAdd(KSUM + bh * D2 + dd, s1);
    atomicAdd(CKSUM + bh * D2 + dd, s2);
  }
  {
    const int d = tid;
    const long ovb = ((long)bh * DD + d) * LL + lblock * LB;
#pragma unroll
    for (int c8 = 0; c8 < 8; ++c8) {
      v8bf a;
#pragma unroll
      for (int i = 0; i < 8; ++i) a[i] = v_s[(c8 * 8 + i) * SV + d];
      *(v8bf*)(Vt + ovb + c8 * 8) = a;
    }
  }
}

// --- K3: z = 1/max(qf . ksum, eps) per (b,h,l), both branches ---------------
__global__ __launch_bounds__(256) void zdenom_kernel(
    const bf16* __restrict__ QF, const bf16* __restrict__ CQF,
    const float* __restrict__ KSUM, const float* __restrict__ CKSUM,
    float* __restrict__ Z1, float* __restrict__ Z2) {
  const long idx = (long)blockIdx.x * 256 + threadIdx.x;  // over B*H*L
  const int bh = (int)(idx >> 11);                        // /L (L=2048)
  const bf16* q1 = QF + idx * D2;
  const bf16* q2 = CQF + idx * D2;
  const float* s1 = KSUM + bh * D2;
  const float* s2 = CKSUM + bh * D2;
  float d1 = 0.f, d2 = 0.f;
#pragma unroll
  for (int c8 = 0; c8 < 16; ++c8) {
    v8bf a = *(const v8bf*)(q1 + c8 * 8);
    v8bf b = *(const v8bf*)(q2 + c8 * 8);
#pragma unroll
    for (int i = 0; i < 8; ++i) {
      const int d = c8 * 8 + i;
      d1 += (float)a[i] * s1[d];
      d2 += (float)b[i] * s2[d];
    }
  }
  Z1[idx] = 1.0f / fmaxf(d1, EPSV);
  Z2[idx] = 1.0f / fmaxf(d2, EPSV);
}

// --- K4: kv = KFt(128,L) @ Vt(64,L)^T per (b,h), BOTH branches fused --------
// One WG per (b,h); 8 waves, each owns one 16-row M tile and all 4 N tiles.
// Vt B-fragments are loaded once per K-step and feed both branch accumulators
// (6 b128-pair loads -> 8 WMMAs per step). Output stored transposed (64x128)
// so it becomes Bt for the attn GEMM.
__global__ __launch_bounds__(256) void kv_gemm_kernel(
    const bf16* __restrict__ KFt, const bf16* __restrict__ CKFt,
    const bf16* __restrict__ Vt,
    bf16* __restrict__ KVT1, bf16* __restrict__ KVT2) {
  const int bh = blockIdx.x;
  const int t = threadIdx.x, wave = t >> 5, lane = t & 31;
  const int r = lane & 15, h = lane >> 4;
  const int cN = lane & 15, moff = (lane >> 4) * 8;
  const bf16* Bt = Vt + (long)bh * DD * LL;
  const bf16* A1 = KFt + (long)bh * D2 * LL + (long)(wave * 16 + r) * LL;
  const bf16* A2 = CKFt + (long)bh * D2 * LL + (long)(wave * 16 + r) * LL;

  v8f acc1[4] = {}, acc2[4] = {};
  for (int kk = 0; kk < LL; kk += 32) {
    v16bf a1 = load_frag(A1 + kk + 8 * h);
    v16bf a2 = load_frag(A2 + kk + 8 * h);
#pragma unroll
    for (int j = 0; j < 4; ++j) {
      v16bf bf = load_frag(Bt + (long)(j * 16 + r) * LL + kk + 8 * h);
      acc1[j] = wmma_bf16(a1, bf, acc1[j]);
      acc2[j] = wmma_bf16(a2, bf, acc2[j]);
    }
  }
  bf16* O1 = KVT1 + (long)bh * DD * D2;
  bf16* O2 = KVT2 + (long)bh * DD * D2;
#pragma unroll
  for (int j = 0; j < 4; ++j)
#pragma unroll
    for (int v = 0; v < 8; ++v) {
      const int m = wave * 16 + moff + v;       // 0..127 (2D index)
      const int n = j * 16 + cN;                // 0..63  (D index)
      O1[n * D2 + m] = (bf16)acc1[j][v];
      O2[n * D2 + m] = (bf16)acc2[j][v];
    }
}

// --- K5: attn = z1*(QF@kv1) + z2*(CQF@kv2), scattered into A4 ---------------
// A4[(l*B + b)*E + h*64 + d] so the final GEMM output lands in d_out flat
// exactly matching the reference's (L,B,E)->(B,L,E) raw reshape.
__global__ __launch_bounds__(256) void attn_gemm_kernel(
    const bf16* __restrict__ QF, const bf16* __restrict__ CQF,
    const bf16* __restrict__ KVT1, const bf16* __restrict__ KVT2,
    const float* __restrict__ Z1, const float* __restrict__ Z2,
    bf16* __restrict__ A4) {
  const int lblock = blockIdx.x;                  // L/128 blocks
  const int bh = blockIdx.y;
  const int b = bh >> 4, hh = bh & 15;
  const int t = threadIdx.x, wave = t >> 5, lane = t & 31;
  const int r = lane & 15, h = lane >> 4;
  const int cN = lane & 15, moff = (lane >> 4) * 8;
  const long lbase = (long)lblock * 128 + wave * 16;

  const bf16* Aq = QF + ((long)bh * LL + lbase + r) * D2;
  const bf16* Ac = CQF + ((long)bh * LL + lbase + r) * D2;
  const bf16* B1 = KVT1 + (long)bh * DD * D2;
  const bf16* B2 = KVT2 + (long)bh * DD * D2;

  v8f acc1[4] = {}, acc2[4] = {};
#pragma unroll
  for (int kk = 0; kk < D2; kk += 32) {
    v16bf a1 = load_frag(Aq + kk + 8 * h);
    v16bf a2 = load_frag(Ac + kk + 8 * h);
#pragma unroll
    for (int j = 0; j < 4; ++j) {
      v16bf b1 = load_frag(B1 + (long)(j * 16 + r) * D2 + kk + 8 * h);
      v16bf b2 = load_frag(B2 + (long)(j * 16 + r) * D2 + kk + 8 * h);
      acc1[j] = wmma_bf16(a1, b1, acc1[j]);
      acc2[j] = wmma_bf16(a2, b2, acc2[j]);
    }
  }
#pragma unroll
  for (int v = 0; v < 8; ++v) {
    const long l = lbase + moff + v;
    const float z1 = Z1[(long)bh * LL + l];
    const float z2 = Z2[(long)bh * LL + l];
    const long rowbase = (l * BB + b) * (long)EE + hh * DD;
#pragma unroll
    for (int j = 0; j < 4; ++j)
      A4[rowbase + j * 16 + cN] = (bf16)(acc1[j][v] * z1 + acc2[j][v] * z2);
  }
}

// ---------------------------------------------------------------------------
extern "C" void kernel_launch(void* const* d_in, const int* in_sizes, int n_in,
                              void* d_out, int out_size, void* d_ws,
                              size_t ws_size, hipStream_t stream) {
  (void)in_sizes; (void)n_in; (void)out_size; (void)ws_size;
  const float* query = (const float*)d_in[0];
  const float* Wq = (const float*)d_in[1]; const float* bq = (const float*)d_in[2];
  const float* Wk = (const float*)d_in[3]; const float* bk = (const float*)d_in[4];
  const float* Wv = (const float*)d_in[5]; const float* bv = (const float*)d_in[6];
  const float* Wo = (const float*)d_in[7]; const float* bo = (const float*)d_in[8];

  // Workspace carve-up (256B aligned regions)
  char* ws = (char*)d_ws;
  size_t off = 0;
  auto carve = [&](size_t bytes) {
    void* p = ws + off;
    off += (bytes + 255) & ~(size_t)255;
    return p;
  };
  constexpr size_t SZ_WT  = (size_t)EE * EE * sizeof(bf16);        // 2 MiB
  constexpr size_t SZ_X   = (size_t)TT * EE * sizeof(bf16);        // 32 MiB
  constexpr size_t SZ_QF  = (size_t)NBH * LL * D2 * sizeof(bf16);  // 64 MiB
  constexpr size_t SZ_VT  = (size_t)NBH * DD * LL * sizeof(bf16);  // 32 MiB
  constexpr size_t SZ_KS  = (size_t)NBH * D2 * sizeof(float);      // 64 KiB
  constexpr size_t SZ_Z   = (size_t)NBH * LL * sizeof(float);      // 1 MiB
  constexpr size_t SZ_KVT = (size_t)NBH * DD * D2 * sizeof(bf16);  // 2 MiB

  bf16* WqT  = (bf16*)carve(SZ_WT);
  bf16* WkT  = (bf16*)carve(SZ_WT);
  bf16* WvT  = (bf16*)carve(SZ_WT);
  bf16* WoT  = (bf16*)carve(SZ_WT);
  bf16* Xbf  = (bf16*)carve(SZ_X);
  bf16* Qbf  = (bf16*)carve(SZ_X);
  bf16* Kbf  = (bf16*)carve(SZ_X);
  bf16* Vbf  = (bf16*)carve(SZ_X);
  bf16* QF   = (bf16*)carve(SZ_QF);
  bf16* CQF  = (bf16*)carve(SZ_QF);
  bf16* KFt  = (bf16*)carve(SZ_QF);
  bf16* CKFt = (bf16*)carve(SZ_QF);
  bf16* Vt   = (bf16*)carve(SZ_VT);
  float* KSUM  = (float*)carve(SZ_KS);
  float* CKSUM = (float*)carve(SZ_KS);
  float* Z1  = (float*)carve(SZ_Z);
  float* Z2  = (float*)carve(SZ_Z);
  bf16* KVT1 = (bf16*)carve(SZ_KVT);
  bf16* KVT2 = (bf16*)carve(SZ_KVT);
  bf16* A4   = Qbf;  // Qbf is dead after featurize; reuse for attn output

  const dim3 bT(32, 8);
  const dim3 gT(EE / 32, EE / 32);
  transpose_w_kernel<<<gT, bT, 0, stream>>>(Wq, WqT);
  transpose_w_kernel<<<gT, bT, 0, stream>>>(Wk, WkT);
  transpose_w_kernel<<<gT, bT, 0, stream>>>(Wv, WvT);
  transpose_w_kernel<<<gT, bT, 0, stream>>>(Wo, WoT);
  convert_bf16_kernel<<<(TT * EE / 4) / 256, 256, 0, stream>>>(query, Xbf);

  // Q/K/V projections (bf16 out, bias fused)
  const dim3 gG(EE / 128, TT / 128);
  gemm_nt_kernel<true><<<gG, 256, 0, stream>>>(Xbf, WqT, bq, Qbf, EE, EE);
  gemm_nt_kernel<true><<<gG, 256, 0, stream>>>(Xbf, WkT, bk, Kbf, EE, EE);
  gemm_nt_kernel<true><<<gG, 256, 0, stream>>>(Xbf, WvT, bv, Vbf, EE, EE);

  // Zero the atomic row-sum accumulators (KSUM,CKSUM are adjacent)
  hipMemsetAsync(KSUM, 0, 2 * SZ_KS, stream);

  featurize_kernel<<<dim3(LL / 64, NBH), 64, 0, stream>>>(
      Qbf, Kbf, Vbf, QF, CQF, KFt, CKFt, Vt, KSUM, CKSUM);

  zdenom_kernel<<<(NBH * LL) / 256, 256, 0, stream>>>(QF, CQF, KSUM, CKSUM,
                                                      Z1, Z2);

  kv_gemm_kernel<<<NBH, 256, 0, stream>>>(KFt, CKFt, Vt, KVT1, KVT2);

  attn_gemm_kernel<<<dim3(LL / 128, NBH), 256, 0, stream>>>(
      QF, CQF, KVT1, KVT2, Z1, Z2, A4);

  // Final projection straight into d_out (fp32, bias fused). Row r = l*B+b of
  // A4 lands at d_out + r*E, which is exactly the reference's raw reshape.
  gemm_nt_kernel<false><<<gG, 256, 0, stream>>>(A4, WoT, bo, (float*)d_out,
                                                EE, EE);
}
